// LLM_MAIN_28630251995423
// MI455X (gfx1250) — matrix operand order repaired and checked by reading, not compiled, not run on hardware
//
#include <hip/hip_runtime.h>
#include <math.h>

#define L_LAYERS 2
#define B_ 4
#define H_ 16
#define KV_ 2
#define G_ 8
#define HD_ 128
#define QKH_ 18
#define HID_ 2048
#define DFF_ 6144
#define VOCAB_ 32000
#define PAST_ 8192
#define NCH_ 32          // flash-decoding chunks per (b,kv)
#define CHUNK_ 256       // keys per chunk (NCH_*CHUNK_ == PAST_)
#define QKVDIM_ 2560     // (QKH+KV)*HD
#define PROW_ 40         // padded pbuf row stride (halfs) -> conflict-free b128

typedef __attribute__((ext_vector_type(16))) _Float16 v16h;
typedef __attribute__((ext_vector_type(8)))  _Float16 v8h;
typedef __attribute__((ext_vector_type(8)))  float    v8f;

// ---- lane broadcast / xor-swizzle helpers (no bpermute index setup) --------
__device__ __forceinline__ float readlane0_f(float x) {
  return __builtin_bit_cast(float, __builtin_amdgcn_readfirstlane(__builtin_bit_cast(int, x)));
}
template <int MASK>
__device__ __forceinline__ float swz_xor(float x) {
  return __builtin_bit_cast(float,
      __builtin_amdgcn_ds_swizzle(__builtin_bit_cast(int, x), (MASK << 10) | 0x1f));
}

// ---- CDNA5 async global->LDS copy (ASYNCcnt-tracked, CDNA5 ISA) ---------
__device__ __forceinline__ void async_copy_b128(unsigned lds, unsigned long long gaddr) {
  asm volatile("global_load_async_to_lds_b128 %0, %1, off" :: "v"(lds), "v"(gaddr) : "memory");
}
__device__ __forceinline__ void wait_async_le16() {
  asm volatile("s_wait_asynccnt 0x10" ::: "memory");
}
__device__ __forceinline__ void wait_async_0() {
  asm volatile("s_wait_asynccnt 0x0" ::: "memory");
}
__device__ __forceinline__ void wait_ds_0() {
  asm volatile("s_wait_dscnt 0x0" ::: "memory");
}

__device__ __forceinline__ v8f wmma_f16(v16h a, v16h b, v8f c) {
  return __builtin_amdgcn_wmma_f32_16x16x32_f16(false, a, false, b, (short)0, c, false, false);
}

// ---------------------------------------------------------------------------
// RMSNorm: y = x * rsqrt(sum(x^2))
// ---------------------------------------------------------------------------
__global__ __launch_bounds__(256)
void rms_kernel(const float* __restrict__ x, float* __restrict__ y, int n) {
  int b = blockIdx.x, tid = threadIdx.x;
  const float* xb = x + (size_t)b * n;
  float s = 0.f;
  for (int i = tid; i < n; i += 256) { float v = xb[i]; s += v * v; }
  __shared__ float red[256];
  red[tid] = s; __syncthreads();
  for (int k = 128; k > 0; k >>= 1) {
    if (tid < k) red[tid] += red[tid + k];
    __syncthreads();
  }
  float inv = rsqrtf(red[0]);
  for (int i = tid; i < n; i += 256) y[(size_t)b * n + i] = xb[i] * inv;
}

// ---------------------------------------------------------------------------
// GEMV: one block per output row, 4 batch accumulators, prefetch ahead.
// ---------------------------------------------------------------------------
__global__ __launch_bounds__(256)
void gemv_kernel(const float* __restrict__ W, const float* __restrict__ xin,
                 const float* __restrict__ bias, const float* __restrict__ resid,
                 float* __restrict__ out, int N, int M) {
  int row = blockIdx.x, tid = threadIdx.x;
  const float4* wr = (const float4*)(W + (size_t)row * N);
  int nv = N >> 2;
  const float4* x0 = (const float4*)xin;
  const float4* x1 = x0 + nv;
  const float4* x2 = x1 + nv;
  const float4* x3 = x2 + nv;
  float a0 = 0.f, a1 = 0.f, a2 = 0.f, a3 = 0.f;
  for (int i = tid; i < nv; i += 256) {
    float4 w = wr[i];
    __builtin_prefetch((const char*)(wr + i) + 16384, 0, 0);
    float4 p;
    p = x0[i]; a0 += w.x * p.x + w.y * p.y + w.z * p.z + w.w * p.w;
    p = x1[i]; a1 += w.x * p.x + w.y * p.y + w.z * p.z + w.w * p.w;
    p = x2[i]; a2 += w.x * p.x + w.y * p.y + w.z * p.z + w.w * p.w;
    p = x3[i]; a3 += w.x * p.x + w.y * p.y + w.z * p.z + w.w * p.w;
  }
  __shared__ float r0[256], r1[256], r2[256], r3[256];
  r0[tid] = a0; r1[tid] = a1; r2[tid] = a2; r3[tid] = a3;
  __syncthreads();
  for (int k = 128; k > 0; k >>= 1) {
    if (tid < k) {
      r0[tid] += r0[tid + k]; r1[tid] += r1[tid + k];
      r2[tid] += r2[tid + k]; r3[tid] += r3[tid + k];
    }
    __syncthreads();
  }
  if (tid == 0) {
    float bv = bias ? bias[row] : 0.f;
    out[(size_t)0 * M + row] = (resid ? resid[(size_t)0 * M + row] : 0.f) + bv + r0[0];
    out[(size_t)1 * M + row] = (resid ? resid[(size_t)1 * M + row] : 0.f) + bv + r1[0];
    out[(size_t)2 * M + row] = (resid ? resid[(size_t)2 * M + row] : 0.f) + bv + r2[0];
    out[(size_t)3 * M + row] = (resid ? resid[(size_t)3 * M + row] : 0.f) + bv + r3[0];
  }
}

// ---------------------------------------------------------------------------
// Per-head RMSNorm + RoPE for the 18 fused q/k heads. One wave per (b, head).
// ---------------------------------------------------------------------------
__global__ __launch_bounds__(32)
void qknorm_rope_kernel(const float* __restrict__ qkv, const float* __restrict__ w,
                        const float* __restrict__ cs, const float* __restrict__ sn,
                        float* __restrict__ outp) {
  int blk = blockIdx.x;
  int b = blk / QKH_, hd = blk % QKH_;
  int lane = threadIdx.x;
  const float* xi = qkv + (size_t)b * QKVDIM_ + (size_t)hd * HD_;
  float v0 = xi[lane], v1 = xi[lane + 32], v2 = xi[lane + 64], v3 = xi[lane + 96];
  float ss = v0 * v0 + v1 * v1 + v2 * v2 + v3 * v3;
  ss += swz_xor<1>(ss); ss += swz_xor<2>(ss); ss += swz_xor<4>(ss);
  ss += swz_xor<8>(ss); ss += swz_xor<16>(ss);
  float inv = rsqrtf(ss);
  const float* wv = w + (size_t)hd * HD_;
  float y0 = v0 * inv * wv[lane];
  float y1 = v1 * inv * wv[lane + 32];
  float y2 = v2 * inv * wv[lane + 64];
  float y3 = v3 * inv * wv[lane + 96];
  float c0 = cs[lane], c1 = cs[lane + 32], c2 = cs[lane + 64], c3 = cs[lane + 96];
  float s0 = sn[lane], s1 = sn[lane + 32], s2 = sn[lane + 64], s3 = sn[lane + 96];
  float* o = outp + ((size_t)b * QKH_ + hd) * HD_;
  o[lane]      = y0 * c0 - y2 * s0;   // rot = -y[d+64]
  o[lane + 32] = y1 * c1 - y3 * s1;
  o[lane + 64] = y2 * c2 + y0 * s2;   // rot = +y[d-64]
  o[lane + 96] = y3 * c3 + y1 * s3;
}

// ---------------------------------------------------------------------------
// Flash-decoding attention chunk: one wave per (b,kv,chunk), 256 past keys,
// processed as 8 pairs of 16-key tiles (K=32 per WMMA macro-step).
//   scores:  8x v_wmma_f32_16x16x32_f16  (A = q f16, rows 8..15 zero;
//            B = contiguous v16h per lane from f16 K cache)
//   P @ V :  8x v_wmma_f32_16x16x32_f16  (A = P via LDS round-trip — the
//            16-bit A layout makes each lane two contiguous 8-half runs;
//            B = V gathered from the async-staged LDS tile; D = ctx accum)
// V tiles (8KB/pair) double-buffered via global_load_async_to_lds_b128 one
// pair ahead; released with s_wait_asynccnt (in-order completion).
// ---------------------------------------------------------------------------
__global__ __launch_bounds__(32)
void attn_chunk_kernel(const _Float16* __restrict__ Kc, const _Float16* __restrict__ Vc,
                       const float* __restrict__ qk,
                       float* __restrict__ pacc, float* __restrict__ pm,
                       float* __restrict__ pl) {
  int blk = blockIdx.x;
  int chunk = blk & (NCH_ - 1);
  int bkv = blk >> 5;              // / NCH_ (==32)
  int b = bkv >> 1, kv = bkv & 1;
  int lane = threadIdx.x;
  int m16 = lane & 15, hh = lane >> 4;

  __shared__ _Float16 vbuf[2][32 * HD_];   // double-buffered V pair (2 x 8KB)
  __shared__ _Float16 pbuf[16 * PROW_];    // P tile [head][key], padded rows
  unsigned ldsv0 = (unsigned)(uintptr_t)&vbuf[0][0] + (unsigned)(lane * 16);
  unsigned ldsv1 = (unsigned)(uintptr_t)&vbuf[1][0] + (unsigned)(lane * 16);

  // Build the 4 q A-fragments (f16) for K-dim tiles of 32.
  const float* qbase = qk + ((size_t)b * QKH_ + (size_t)kv * G_) * HD_;
  v16h afr[4];
#pragma unroll
  for (int kt = 0; kt < 4; ++kt) {
#pragma unroll
    for (int e = 0; e < 16; ++e) {
      int j = e >> 1;
      int kidx = (hh << 4) + e;
      float qv = (m16 < G_) ? qbase[(size_t)m16 * HD_ + kt * 32 + kidx] : 0.f;
      afr[kt][e] = (_Float16)qv;
    }
  }

  size_t cb = (((size_t)b * KV_ + kv) * PAST_ + (size_t)chunk * CHUNK_) * HD_;
  const _Float16* Kb = Kc + cb;
  unsigned long long vg = (unsigned long long)(uintptr_t)(Vc + cb) + (unsigned)(lane * 16);

  // Prefetch V pair 0 into buffer 0 (16 x 512B slabs across the wave).
#pragma unroll
  for (int i = 0; i < 16; ++i)
    async_copy_b128(ldsv0 + i * 512, vg + i * 512);

  float mrun[G_], lrun[G_];
  v8f ctxa[8];
#pragma unroll
  for (int r = 0; r < G_; ++r) { mrun[r] = -1e30f; lrun[r] = 0.f; }
#pragma unroll
  for (int nt = 0; nt < 8; ++nt) ctxa[nt] = (v8f){};

  for (int tp = 0; tp < CHUNK_ / 32; ++tp) {
    int cur = tp & 1;
    if (tp + 1 < CHUNK_ / 32) {
      unsigned nb = (cur ? ldsv0 : ldsv1);
      unsigned long long ng = vg + (unsigned long long)(tp + 1) * (32 * HD_ * 2);
#pragma unroll
      for (int i = 0; i < 16; ++i)
        async_copy_b128(nb + i * 512, ng + i * 512);
    }

    // ---- scores for two 16-key tiles ----
    const _Float16* krow0 = Kb + (size_t)(tp * 32 + m16) * HD_;
    const _Float16* krow1 = krow0 + (size_t)16 * HD_;
    v8f c0 = (v8f){}, c1 = (v8f){};
#pragma unroll
    for (int kk = 0; kk < 4; ++kk) {
      v16h bf = *(const v16h*)(krow0 + kk * 32 + hh * 16);
      c0 = wmma_f16(afr[kk], bf, c0);
    }
#pragma unroll
    for (int kk = 0; kk < 4; ++kk) {
      v16h bf = *(const v16h*)(krow1 + kk * 32 + hh * 16);
      c1 = wmma_f16(afr[kk], bf, c1);
    }

    // ---- streaming softmax over the 32-key pair ----
    float scv[G_];
#pragma unroll
    for (int r = 0; r < G_; ++r) {
      float v = fmaxf(c0[r], c1[r]);
      v = fmaxf(v, swz_xor<1>(v)); v = fmaxf(v, swz_xor<2>(v));
      v = fmaxf(v, swz_xor<4>(v)); v = fmaxf(v, swz_xor<8>(v));
      float tm = readlane0_f(v);
      float mn = fmaxf(mrun[r], tm);
      scv[r] = __expf(mrun[r] - mn);
      mrun[r] = mn;
      float p0 = __expf(c0[r] - mn);
      float p1 = __expf(c1[r] - mn);
      float srow = p0 + p1;
      srow += swz_xor<1>(srow); srow += swz_xor<2>(srow);
      srow += swz_xor<4>(srow); srow += swz_xor<8>(srow);
      lrun[r] = lrun[r] * scv[r] + readlane0_f(srow);
      c0[r] = p0; c1[r] = p1;
    }

    // ---- P -> LDS (head-major, keys contiguous) ----
#pragma unroll
    for (int r = 0; r < G_; ++r) {
      pbuf[(r + 8 * hh) * PROW_ + m16]      = (_Float16)c0[r];
      pbuf[(r + 8 * hh) * PROW_ + 16 + m16] = (_Float16)c1[r];
    }
    wait_ds_0();

    // ---- P A-fragment: two contiguous 8-half runs per lane ----
    v16h pa;
    {
      v8h lo = *(const v8h*)(&pbuf[m16 * PROW_ + hh * 8]);
      v8h hi = *(const v8h*)(&pbuf[m16 * PROW_ + 16 + hh * 8]);
#pragma unroll
      for (int i = 0; i < 8; ++i) { pa[i] = lo[i]; pa[8 + i] = hi[i]; }
    }

    // ---- V B-fragments from the async-staged LDS tile + PV WMMAs ----
    if (tp + 1 < CHUNK_ / 32) wait_async_le16();
    else                      wait_async_0();
    const _Float16* vb = &vbuf[cur][0];
#pragma unroll
    for (int nt = 0; nt < 8; ++nt) {
      v16h vf;
#pragma unroll
      for (int e = 0; e < 16; ++e)
        vf[e] = vb[(size_t)(((e >> 3) << 4) + (hh << 3) + (e & 7)) * HD_ + nt * 16 + m16];
#pragma unroll
      for (int r = 0; r < G_; ++r) ctxa[nt][r] *= scv[r];   // flash rescale
      ctxa[nt] = wmma_f16(pa, vf, ctxa[nt]);
    }
  }

  // ---- write partials: heads 0..7 live on lanes 0..15 (h==0 half) ----
  if (hh == 0) {
    float* po = pacc + (size_t)blk * G_ * HD_;
#pragma unroll
    for (int nt = 0; nt < 8; ++nt)
#pragma unroll
      for (int r = 0; r < G_; ++r)
        po[(size_t)r * HD_ + nt * 16 + m16] = ctxa[nt][r];
  }
  if (lane == 0) {
#pragma unroll
    for (int r = 0; r < G_; ++r) {
      pm[(size_t)blk * G_ + r] = mrun[r];
      pl[(size_t)blk * G_ + r] = lrun[r];
    }
  }
}

// ---------------------------------------------------------------------------
// Combine chunk partials + current-token K/V contribution, normalize.
// ---------------------------------------------------------------------------
__global__ __launch_bounds__(32)
void attn_combine_kernel(const float* __restrict__ qk, const float* __restrict__ qkv,
                         const float* __restrict__ pacc, const float* __restrict__ pm,
                         const float* __restrict__ pl, float* __restrict__ ctx) {
  int bkv = blockIdx.x;
  int b = bkv >> 1, kv = bkv & 1;
  int lane = threadIdx.x;
  int d = lane * 4;
  const float* qbase = qk + ((size_t)b * QKH_ + (size_t)kv * G_) * HD_;
  const float* knew  = qk + ((size_t)b * QKH_ + H_ + kv) * HD_;
  float k0 = knew[d], k1 = knew[d + 1], k2 = knew[d + 2], k3 = knew[d + 3];
  const float* vn = qkv + (size_t)b * QKVDIM_ + (size_t)QKH_ * HD_ + (size_t)kv * HD_;
  float v0 = vn[d], v1 = vn[d + 1], v2 = vn[d + 2], v3 = vn[d + 3];

  float snew[G_];
#pragma unroll
  for (int r = 0; r < G_; ++r) {
    const float* qr = qbase + (size_t)r * HD_;
    float s = qr[d] * k0 + qr[d + 1] * k1 + qr[d + 2] * k2 + qr[d + 3] * k3;
    s += swz_xor<1>(s); s += swz_xor<2>(s); s += swz_xor<4>(s);
    s += swz_xor<8>(s); s += swz_xor<16>(s);
    snew[r] = s;
  }
  size_t base = (size_t)bkv * NCH_ * G_;
#pragma unroll
  for (int r = 0; r < G_; ++r) {
    float M = snew[r];
    for (int i = 0; i < NCH_; ++i) M = fmaxf(M, pm[base + (size_t)i * G_ + r]);
    float wnew = __expf(snew[r] - M);
    float Ltot = wnew;
    float ax = wnew * v0, ay = wnew * v1, az = wnew * v2, aw = wnew * v3;
    for (int i = 0; i < NCH_; ++i) {
      float mi = pm[base + (size_t)i * G_ + r];
      float li = pl[base + (size_t)i * G_ + r];
      float w = __expf(mi - M);
      Ltot += li * w;
      const float* pa = pacc + (base + (size_t)i * G_ + r) * HD_ + d;
      ax += w * pa[0]; ay += w * pa[1]; az += w * pa[2]; aw += w * pa[3];
    }
    float inv = 1.f / Ltot;
    float* o = ctx + (size_t)b * (H_ * HD_) + (size_t)(kv * G_ + r) * HD_ + d;
    o[0] = ax * inv; o[1] = ay * inv; o[2] = az * inv; o[3] = aw * inv;
  }
}

// ---------------------------------------------------------------------------
// SwiGLU: h = silu(gate) * up
// ---------------------------------------------------------------------------
__global__ __launch_bounds__(256)
void silu_kernel(const float* __restrict__ gu, float* __restrict__ hh) {
  int idx = blockIdx.x * 256 + threadIdx.x;
  if (idx >= B_ * DFF_) return;
  int b = idx / DFF_, f = idx - b * DFF_;
  float g = gu[(size_t)b * 2 * DFF_ + f];
  float u = gu[(size_t)b * 2 * DFF_ + DFF_ + f];
  hh[idx] = (g / (1.f + __expf(-g))) * u;
}

// ---------------------------------------------------------------------------
extern "C" void kernel_launch(void* const* d_in, const int* in_sizes, int n_in,
                              void* d_out, int out_size, void* d_ws, size_t ws_size,
                              hipStream_t stream) {
  (void)in_sizes; (void)n_in; (void)out_size; (void)ws_size;
  const float*    hidden    = (const float*)d_in[0];
  const float*    cosp      = (const float*)d_in[1];
  const float*    sinp      = (const float*)d_in[2];
  const _Float16* past_k    = (const _Float16*)d_in[3];
  const _Float16* past_v    = (const _Float16*)d_in[4];
  const float*    qkv_w     = (const float*)d_in[5];
  const float*    qkv_b     = (const float*)d_in[6];
  const float*    qk_norm_w = (const float*)d_in[7];
  const float*    o_w       = (const float*)d_in[8];
  const float*    gate_up_w = (const float*)d_in[9];
  const float*    down_w    = (const float*)d_in[10];
  const float*    lm_w      = (const float*)d_in[11];
  float* out = (float*)d_out;

  float* ws   = (float*)d_ws;
  float* X    = ws;                                    // [B][HID]
  float* XN   = X    + (size_t)B_ * HID_;              // [B][HID]
  float* QKV  = XN   + (size_t)B_ * HID_;              // [B][2560]
  float* QKR  = QKV  + (size_t)B_ * QKVDIM_;           // [B][QKH][HD]
  float* PACC = QKR  + (size_t)B_ * QKH_ * HD_;        // [B*KV*NCH][G][HD]
  float* PM   = PACC + (size_t)B_ * KV_ * NCH_ * G_ * HD_;
  float* PL   = PM   + (size_t)B_ * KV_ * NCH_ * G_;
  float* CTX  = PL   + (size_t)B_ * KV_ * NCH_ * G_;   // [B][H*HD]
  float* GU   = CTX  + (size_t)B_ * H_ * HD_;          // [B][2*DFF]
  float* HH   = GU   + (size_t)B_ * 2 * DFF_;          // [B][DFF]

  hipMemcpyAsync(X, hidden, sizeof(float) * B_ * HID_, hipMemcpyDeviceToDevice, stream);

  for (int l = 0; l < L_LAYERS; ++l) {
    rms_kernel<<<B_, 256, 0, stream>>>(X, XN, HID_);
    gemv_kernel<<<QKVDIM_, 256, 0, stream>>>(qkv_w + (size_t)l * QKVDIM_ * HID_, XN,
                                             qkv_b + (size_t)l * QKVDIM_, nullptr,
                                             QKV, HID_, QKVDIM_);
    qknorm_rope_kernel<<<B_ * QKH_, 32, 0, stream>>>(
        QKV, qk_norm_w + (size_t)l * QKH_ * HD_, cosp, sinp, QKR);
    attn_chunk_kernel<<<B_ * KV_ * NCH_, 32, 0, stream>>>(
        past_k + (size_t)l * B_ * KV_ * PAST_ * HD_,
        past_v + (size_t)l * B_ * KV_ * PAST_ * HD_,
        QKR, PACC, PM, PL);
    attn_combine_kernel<<<B_ * KV_, 32, 0, stream>>>(QKR, QKV, PACC, PM, PL, CTX);
    gemv_kernel<<<HID_, 256, 0, stream>>>(o_w + (size_t)l * HID_ * H_ * HD_, CTX,
                                          nullptr, X, X, H_ * HD_, HID_);
    rms_kernel<<<B_, 256, 0, stream>>>(X, XN, HID_);
    gemv_kernel<<<2 * DFF_, 256, 0, stream>>>(gate_up_w + (size_t)l * 2 * DFF_ * HID_, XN,
                                              nullptr, nullptr, GU, HID_, 2 * DFF_);
    silu_kernel<<<(B_ * DFF_ + 255) / 256, 256, 0, stream>>>(GU, HH);
    gemv_kernel<<<HID_, 256, 0, stream>>>(down_w + (size_t)l * HID_ * DFF_, HH,
                                          nullptr, X, X, DFF_, HID_);
  }
  rms_kernel<<<B_, 256, 0, stream>>>(X, XN, HID_);
  gemv_kernel<<<VOCAB_, 256, 0, stream>>>(lm_w, XN, nullptr, nullptr, out, HID_, VOCAB_);
}
